// AlternatingForecastModel_16561393893535
// MI455X (gfx1250) — 8-bit matrix operand order repaired and checked by reading, compiled, not run on hardware
//
#include <hip/hip_runtime.h>

#define BB 128
#define SS 2048
#define FF 16
#define HH 256

#define HSTRIDE 272
#define CSTRIDE 80

typedef __attribute__((ext_vector_type(8))) int          v8i;
typedef __attribute__((ext_vector_type(8))) float        v8f;
typedef __attribute__((ext_vector_type(2))) unsigned int u32x2;
typedef __attribute__((ext_vector_type(4))) unsigned int u32x4;


__device__ __forceinline__ unsigned f2e4m3_sw(float x) {
  union { float f; unsigned u; } v; v.f = x;
  unsigned s = (v.u >> 31) << 7;
  v.u &= 0x7FFFFFFFu;
  if (v.f != v.f) return s | 0x7Fu;
  if (v.f >= 448.0f) return s | 0x7Eu;
  if (v.f < 0.0009765625f) return s;
  int e = (int)((v.u >> 23) & 0xFF) - 127;
  if (e < -6) {
    int m = (int)(v.f * 512.0f + 0.5f);
    if (m >= 8) return s | 0x08u;
    return s | (unsigned)m;
  }
  unsigned r = v.u + (1u << 19);
  int e2 = (int)((r >> 23) & 0xFF) - 127;
  if (e2 > 8) return s | 0x7Eu;
  unsigned m3 = (r >> 20) & 0x7u;
  return s | ((unsigned)(e2 + 7) << 3) | m3;
}

__device__ __forceinline__ unsigned cvt1_fp8(float a) {
#if __has_builtin(__builtin_amdgcn_cvt_pk_fp8_f32)
  return (unsigned)__builtin_amdgcn_cvt_pk_fp8_f32(a, 0.0f, 0, false);
#else
  return f2e4m3_sw(a);
#endif
}

__device__ __forceinline__ unsigned pk4_fp8(float a, float b, float c, float d) {
#if __has_builtin(__builtin_amdgcn_cvt_pk_fp8_f32)
  int v = __builtin_amdgcn_cvt_pk_fp8_f32(a, b, 0, false);
  v = __builtin_amdgcn_cvt_pk_fp8_f32(c, d, v, true);
  return (unsigned)v;
#else
  return f2e4m3_sw(a) | (f2e4m3_sw(b) << 8) | (f2e4m3_sw(c) << 16) | (f2e4m3_sw(d) << 24);
#endif
}

__device__ __forceinline__ float fast_rcp(float x) {
#if __has_builtin(__builtin_amdgcn_rcpf)
  return __builtin_amdgcn_rcpf(x);
#else
  return 1.0f / x;
#endif
}

__device__ __forceinline__ float sigmoidf_fast(float x) {
  return fast_rcp(1.0f + __expf(-x));
}
__device__ __forceinline__ float tanhf_fast(float x) {
  x = fminf(15.0f, fmaxf(-15.0f, x));
  float e = __expf(2.0f * x);
  return (e - 1.0f) * fast_rcp(e + 1.0f);
}

__device__ __forceinline__ v8i load_a_frag(const unsigned char* rowp, int hi) {
  union { u32x2 q[4]; v8i v; } u;
  const int kb = hi * 16;
  u.q[0] = *(const u32x2*)(rowp + kb);
  u.q[1] = *(const u32x2*)(rowp + kb + 8);
  u.q[2] = *(const u32x2*)(rowp + kb + 32);
  u.q[3] = *(const u32x2*)(rowp + kb + 40);
  return u.v;
}

__global__ void anom_scan(const float* __restrict__ x, float* __restrict__ anom) {
  const int t = blockIdx.x;
  __shared__ int flag;
  if (threadIdx.x == 0) flag = 0;
  __syncthreads();
  float v = x[(size_t)threadIdx.x * (SS * FF) + (size_t)t * FF + (FF - 1)];
  if (v != 0.0f) atomicOr(&flag, 1);
  __syncthreads();
  if (threadIdx.x == 0) anom[t] = flag ? 1.0f : 0.0f;
}

__global__ __launch_bounds__(512, 1)
void lstm_persist(const float* __restrict__ x,
                  const float* __restrict__ W_ih, const float* __restrict__ b_ih,
                  const float* __restrict__ W_hh, const float* __restrict__ b_hh,
                  const float* __restrict__ W_out, const float* __restrict__ b_out,
                  const float* __restrict__ anom_g,
                  float* __restrict__ out) {
  __shared__ __align__(16) unsigned char h_lds[2][16 * HSTRIDE];
  __shared__ __align__(16) unsigned char cur_lds[16 * CSTRIDE];
  __shared__ __align__(16) unsigned int  wihF[4][16][32][4];
  __shared__ float out_acc[2][16];
  __shared__ float anom_s[SS];

  const int tid  = threadIdx.x;
  const int wave = tid >> 5;
  const int lane = tid & 31;
  const int lrow = lane & 15;
  const int hi   = lane >> 4;
  const int b0   = blockIdx.x * 16;
  const int colw = wave * 16 + lrow;
  const int Mrow = hi * 8;

  for (int i = tid; i < 2 * 16 * HSTRIDE; i += 512) ((unsigned char*)h_lds)[i] = 0;
  for (int i = tid; i < 16 * CSTRIDE; i += 512) cur_lds[i] = 0;
  for (int i = tid; i < SS; i += 512) anom_s[i] = anom_g[i];
  if (tid < 32) out_acc[tid >> 4][tid & 15] = 0.0f;
  if (tid < 16)
    cur_lds[tid * CSTRIDE + (FF + 1)] = (unsigned char)cvt1_fp8(1.0f);

  for (int g = 0; g < 4; ++g) {
    const int n = g * HH + colw;
    const float* wi = W_ih + (size_t)n * (FF + 1);
    u32x4 q;
    if (hi == 0) {
      q[0] = pk4_fp8(wi[0],  wi[1],  wi[2],  wi[3]);
      q[1] = pk4_fp8(wi[4],  wi[5],  wi[6],  wi[7]);
      q[2] = pk4_fp8(wi[8],  wi[9],  wi[10], wi[11]);
      q[3] = pk4_fp8(wi[12], wi[13], wi[14], wi[15]);
    } else {
      q[0] = (cvt1_fp8(wi[FF]) & 0xFFu) |
             ((cvt1_fp8(b_ih[n] + b_hh[n]) & 0xFFu) << 8);
      q[1] = 0; q[2] = 0; q[3] = 0;
    }
    *(u32x4*)&wihF[g][wave][lane][0] = q;
  }

  v8i whh[4][4];
#pragma unroll
  for (int g = 0; g < 4; ++g) {
    const float* wr = W_hh + (size_t)(g * HH + colw) * HH;
#pragma unroll
    for (int kc = 0; kc < 4; ++kc) {
#pragma unroll
      for (int d = 0; d < 8; ++d) {
        int kb = kc * 64 + (d >> 2) * 32 + hi * 16 + (d & 3) * 4;
        whh[g][kc][d] = (int)pk4_fp8(wr[kb], wr[kb + 1], wr[kb + 2], wr[kb + 3]);
      }
    }
  }
  const float woutv = W_out[colw];
  const float boutv = b_out[0];

  float c_st[8], lg_st[8], h_st[8];
#pragma unroll
  for (int r = 0; r < 8; ++r) { c_st[r] = 0.0f; lg_st[r] = 0.0f; h_st[r] = 0.0f; }

  __syncthreads();

  for (int t = 0; t < SS; ++t) {
    const int cb = t & 1;
    const int pb = cb ^ 1;

    if (t > 0 && wave == 0 && lane < 16)
      out[(size_t)(b0 + lane) * SS + (t - 1)] = out_acc[pb][lane] + boutv;
    if (tid < 16) out_acc[cb][tid] = 0.0f;
    {
      const int row = tid >> 5;
      const int ft  = lane;
      const bool pred = (t >= 672) && (t < 1344);
      if (ft <= FF) {
        const float* xr = x + (size_t)(b0 + row) * (SS * FF) + (size_t)t * FF;
        float v;
        if (ft == 0)      v = pred ? (out_acc[pb][row] + boutv) : xr[0];
        else if (ft < FF) v = xr[ft];
        else              v = pred ? 1.0f : 0.0f;
        cur_lds[row * CSTRIDE + ft] = (unsigned char)cvt1_fp8(v);
      }
      if (ft < FF && t + 1 < SS)
        __builtin_prefetch(x + (size_t)(b0 + row) * (SS * FF) + (size_t)(t + 1) * FF + ft, 0, 1);
    }
    __syncthreads();

    v8f acc[4];
    {
      v8i ax = load_a_frag(cur_lds + lrow * CSTRIDE, hi);
#pragma unroll
      for (int g = 0; g < 4; ++g) {
        u32x4 q = *(const u32x4*)&wihF[g][wave][lane][0];
        v8i bx = {};
        bx[0] = (int)q[0]; bx[1] = (int)q[1]; bx[2] = (int)q[2]; bx[3] = (int)q[3];
        v8f z = {};
        acc[g] = __builtin_amdgcn_wmma_f32_16x16x64_fp8_fp8(
            ax, bx, (short)0, z, false, false);
      }
#pragma unroll
      for (int kc = 0; kc < 4; ++kc) {
        v8i ah = load_a_frag(&h_lds[cb][lrow * HSTRIDE + kc * 64], hi);
#pragma unroll
        for (int g = 0; g < 4; ++g)
          acc[g] = __builtin_amdgcn_wmma_f32_16x16x64_fp8_fp8(
              ah, whh[g][kc], (short)0, acc[g], false, false);
      }
    }

    const float isan = anom_s[t];
    unsigned char* hnext = h_lds[cb ^ 1];
    float p_red[8];
#pragma unroll
    for (int r = 0; r < 8; ++r) {
      float iv = sigmoidf_fast(acc[0][r]);
      float fv = sigmoidf_fast(acc[1][r]);
      float gv = tanhf_fast(acc[2][r]);
      float ov = sigmoidf_fast(acc[3][r]);
      float nc = fv * c_st[r] + iv * gv;
      float nh = ov * tanhf_fast(nc);
      float blend = lg_st[r] * 0.7f + nc * 0.3f;
      float ca = c_st[r] * 0.9f + blend * 0.1f;
      float cn = (isan != 0.0f) ? ca : nc;
      lg_st[r] = (isan != 0.0f) ? lg_st[r] : cn;
      c_st[r] = cn;
      h_st[r] = nh;
      hnext[(Mrow + r) * HSTRIDE + colw] = (unsigned char)cvt1_fp8(nh);

      float p = nh * woutv;
#pragma unroll
      for (int md = 1; md < 16; md <<= 1) p += __shfl_xor(p, md, 32);
      p_red[r] = p;
    }
    if (lrow == 0) {
#pragma unroll
      for (int r = 0; r < 8; ++r) atomicAdd(&out_acc[cb][Mrow + r], p_red[r]);
    }
    __syncthreads();
  }

  if (wave == 0 && lane < 16)
    out[(size_t)(b0 + lane) * SS + (SS - 1)] = out_acc[(SS - 1) & 1][lane] + boutv;
  float* hfin = out + (size_t)BB * SS;
  float* cfin = hfin + (size_t)BB * HH;
#pragma unroll
  for (int r = 0; r < 8; ++r) {
    hfin[(size_t)(b0 + Mrow + r) * HH + colw] = h_st[r];
    cfin[(size_t)(b0 + Mrow + r) * HH + colw] = c_st[r];
  }
}

extern "C" void kernel_launch(void* const* d_in, const int* in_sizes, int n_in,
                              void* d_out, int out_size, void* d_ws, size_t ws_size,
                              hipStream_t stream) {
  const float* x     = (const float*)d_in[0];
  const float* W_ih  = (const float*)d_in[1];
  const float* b_ih  = (const float*)d_in[2];
  const float* W_hh  = (const float*)d_in[3];
  const float* b_hh  = (const float*)d_in[4];
  const float* W_out = (const float*)d_in[5];
  const float* b_out = (const float*)d_in[6];
  float* out  = (float*)d_out;
  float* anom = (float*)d_ws;

  anom_scan<<<SS, BB, 0, stream>>>(x, anom);
  lstm_persist<<<BB / 16, 512, 0, stream>>>(x, W_ih, b_ih, W_hh, b_hh,
                                            W_out, b_out, anom, out);
}
